// _GraphConvolutionLayer_88802743812239
// MI455X (gfx1250) — matrix operand order repaired and checked by reading, not compiled, not run on hardware
//
#include <hip/hip_runtime.h>
#include <cmath>

// ---------------------------------------------------------------------------
// gfx1250 (CDNA5, wave32) implementation of the 5-MHA graph-attention block.
// All GEMMs run on v_wmma_f32_16x16x32_bf16 (fp32 accumulate).
// ---------------------------------------------------------------------------

typedef __attribute__((ext_vector_type(16))) __bf16 v16bf;
typedef __attribute__((ext_vector_type(8)))  __bf16 v8bf;
typedef __attribute__((ext_vector_type(8)))  float  v8f;

__device__ __forceinline__ unsigned short f32_to_bf16_rte(float f) {
  unsigned int u = __float_as_uint(f);
  u += 0x7FFFu + ((u >> 16) & 1u);          // round-to-nearest-even
  return (unsigned short)(u >> 16);
}

// ---------------------------------------------------------------------------
// Generic WMMA GEMM:  C[z][m][n] = epi( scale * sum_k A[z][m][k]*B[z][k][n] + bias[n] )
//   A (bf16):  !A_TRANS : A + z*aB + m*lda + k
//               A_TRANS : A + z*aB + k*lda + m            (P^T operand)
//   B (bf16):  B_MODE 0 : B + z*bB + k*ldb + n            (weights stored [K,N])
//              B_MODE 1 : B + z*bB + n*ldb + k            (K-matrix rows as columns, QK^T)
// Tile: 128x128, TK=32. 256 threads = 8 waves in a 4x2 grid; wave tile 32x64.
// ---------------------------------------------------------------------------
constexpr int TM = 128, TN = 128, TK = 32;
constexpr int LSTR = TK + 8;                 // padded LDS row stride (ushorts)

template<bool A_TRANS, int B_MODE, bool OUT_BF16, bool DO_ELU, bool HAS_BIAS>
__global__ __launch_bounds__(256)
void wmma_gemm(const unsigned short* __restrict__ A, long long aB, int lda,
               const unsigned short* __restrict__ B, long long bB, int ldb,
               void* __restrict__ Cv, long long cB, int ldc,
               const float* __restrict__ bias,
               int M, int N, int K, float scale)
{
  __shared__ unsigned short As[TM * LSTR];
  __shared__ unsigned short Bs[TN * LSTR];

  const int tid  = threadIdx.x;
  const int lane = tid & 31;
  const int wave = tid >> 5;
  const int wm   = wave >> 1;                // 0..3 : 32-row slabs
  const int wn   = wave &  1;                // 0..1 : 64-col slabs
  const int m0   = blockIdx.y * TM;
  const int n0   = blockIdx.x * TN;
  const int half = lane >> 4;                // lane group (K-interleave select)
  const int lr   = lane & 15;

  A += (long long)blockIdx.z * aB;
  B += (long long)blockIdx.z * bB;

  v8f acc[2][4];
  #pragma unroll
  for (int i = 0; i < 2; ++i)
    #pragma unroll
    for (int j = 0; j < 4; ++j)
      acc[i][j] = (v8f){0.f,0.f,0.f,0.f,0.f,0.f,0.f,0.f};

  for (int k0 = 0; k0 < K; k0 += TK) {
    // ---- stage A tile (TM x TK) into As[m][k]
    if constexpr (!A_TRANS) {
      const int row = tid >> 1, cs = (tid & 1) * 16;
      const unsigned short* g = A + (long long)(m0 + row) * lda + (k0 + cs);
      uint4 d0 = *(const uint4*)g;
      uint4 d1 = *(const uint4*)(g + 8);
      *(uint4*)&As[row * LSTR + cs]     = d0;
      *(uint4*)&As[row * LSTR + cs + 8] = d1;
    } else {                                  // read along m (coalesced), scatter-transpose
      const int kr = tid >> 3, ms = (tid & 7) * 16;
      const unsigned short* g = A + (long long)(k0 + kr) * lda + (m0 + ms);
      union { uint4 q[2]; unsigned short s[16]; } u;
      u.q[0] = *(const uint4*)g;
      u.q[1] = *(const uint4*)(g + 8);
      #pragma unroll
      for (int e = 0; e < 16; ++e) As[(ms + e) * LSTR + kr] = u.s[e];
    }
    // ---- stage B tile into Bs[n][k]  (B^T layout so fragment reads are contiguous)
    if constexpr (B_MODE == 0) {
      const int kr = tid >> 3, ns = (tid & 7) * 16;
      const unsigned short* g = B + (long long)(k0 + kr) * ldb + (n0 + ns);
      union { uint4 q[2]; unsigned short s[16]; } u;
      u.q[0] = *(const uint4*)g;
      u.q[1] = *(const uint4*)(g + 8);
      #pragma unroll
      for (int e = 0; e < 16; ++e) Bs[(ns + e) * LSTR + kr] = u.s[e];
    } else {
      const int row = tid >> 1, cs = (tid & 1) * 16;
      const unsigned short* g = B + (long long)(n0 + row) * ldb + (k0 + cs);
      uint4 d0 = *(const uint4*)g;
      uint4 d1 = *(const uint4*)(g + 8);
      *(uint4*)&Bs[row * LSTR + cs]     = d0;
      *(uint4*)&Bs[row * LSTR + cs + 8] = d1;
    }
    __syncthreads();

    // ---- fragments (per documented 16-bit A/B VGPR layouts) + 8 WMMAs
    v16bf afrag[2], bfrag[4];
    #pragma unroll
    for (int i = 0; i < 2; ++i) {
      // A 16x32: lanes 0-15 hold K 0-7 (elems 0-7) & 16-23 (elems 8-15); lanes 16-31: +8
      const unsigned short* p = &As[(wm * 32 + i * 16 + lr) * LSTR + half * 8];
      union { v16bf v; v8bf h[2]; } u;
      u.h[0] = *(const v8bf*)p;
      u.h[1] = *(const v8bf*)(p + 16);
      afrag[i] = u.v;
    }
    #pragma unroll
    for (int j = 0; j < 4; ++j) {
      // B 32x16: lanes 0-15 hold K 0-7 (elems 0-7) & 16-23 (elems 8-15); lanes 16-31: +8 (col = lane&15)
      const unsigned short* p = &Bs[(wn * 64 + j * 16 + lr) * LSTR + half * 8];
      union { v16bf v; v8bf h[2]; } u;
      u.h[0] = *(const v8bf*)p;
      u.h[1] = *(const v8bf*)(p + 16);
      bfrag[j] = u.v;
    }
    #pragma unroll
    for (int i = 0; i < 2; ++i)
      #pragma unroll
      for (int j = 0; j < 4; ++j)
        acc[i][j] = __builtin_amdgcn_wmma_f32_16x16x32_bf16(
            false, afrag[i], false, bfrag[j], (short)0, acc[i][j], false, false);
    __syncthreads();
  }

  // ---- fused epilogue: scale, bias, elu, f32/bf16 store
  const long long cOff = (long long)blockIdx.z * cB;
  #pragma unroll
  for (int i = 0; i < 2; ++i) {
    const int mb = m0 + wm * 32 + i * 16 + half * 8;   // C/D layout: VGPR r -> row r or 8+r
    #pragma unroll
    for (int j = 0; j < 4; ++j) {
      const int n = n0 + wn * 64 + j * 16 + lr;
      float bv = 0.f;
      if constexpr (HAS_BIAS) bv = bias[n];
      #pragma unroll
      for (int r = 0; r < 8; ++r) {
        float v = acc[i][j][r] * scale + bv;
        if constexpr (DO_ELU) v = (v > 0.f) ? v : (__expf(v) - 1.f);
        const long long idx = cOff + (long long)(mb + r) * ldc + n;
        if constexpr (OUT_BF16) ((unsigned short*)Cv)[idx] = f32_to_bf16_rte(v);
        else                    ((float*)Cv)[idx] = v;
      }
    }
  }
}

// ---------------------------------------------------------------------------
// Row softmax (one block per row), optional diagonal mask (mask = 1 - eye),
// output written as bf16 probabilities (feeds the next WMMA GEMMs).
// ---------------------------------------------------------------------------
template<bool MASK_DIAG>
__global__ __launch_bounds__(256)
void softmax_rows(const float* __restrict__ S, unsigned short* __restrict__ P,
                  int ncols, int rowsPerHead)
{
  const int row = blockIdx.x;
  const float* s = S + (long long)row * ncols;
  unsigned short* p = P + (long long)row * ncols;
  const int diag = MASK_DIAG ? (row % rowsPerHead) : -1;
  const int tid = threadIdx.x;
  __shared__ float red[8];

  float mx = -3.0e38f;
  for (int c = tid; c < ncols; c += 256) {
    float v = (c == diag) ? -1.0e9f : s[c];
    mx = fmaxf(mx, v);
  }
  #pragma unroll
  for (int o = 16; o > 0; o >>= 1) mx = fmaxf(mx, __shfl_down(mx, o, 32));
  if ((tid & 31) == 0) red[tid >> 5] = mx;
  __syncthreads();
  if (tid == 0) { float v = red[0]; for (int w = 1; w < 8; ++w) v = fmaxf(v, red[w]); red[0] = v; }
  __syncthreads();
  const float rowmax = red[0];
  __syncthreads();

  float sum = 0.f;
  for (int c = tid; c < ncols; c += 256) {
    float v = (c == diag) ? -1.0e9f : s[c];
    sum += __expf(v - rowmax);
  }
  #pragma unroll
  for (int o = 16; o > 0; o >>= 1) sum += __shfl_down(sum, o, 32);
  if ((tid & 31) == 0) red[tid >> 5] = sum;
  __syncthreads();
  if (tid == 0) { float v = 0.f; for (int w = 0; w < 8; ++w) v += red[w]; red[0] = v; }
  __syncthreads();
  const float inv = 1.0f / red[0];
  for (int c = tid; c < ncols; c += 256) {
    float v = (c == diag) ? -1.0e9f : s[c];
    p[c] = f32_to_bf16_rte(__expf(v - rowmax) * inv);
  }
}

// ---------------------------------------------------------------------------
// Elementwise helpers
// ---------------------------------------------------------------------------
__global__ __launch_bounds__(256)
void cvt_f32_bf16(const float* __restrict__ in, unsigned short* __restrict__ out, int n) {
  int i = blockIdx.x * 256 + threadIdx.x;
  if (i < n) out[i] = f32_to_bf16_rte(in[i]);
}

// sum 6 heads (elu already applied per head) into a column slice of C
__global__ __launch_bounds__(256)
void head_sum6(const float* __restrict__ T, float* __restrict__ C,
               int N, int dk, int ldc, int colOff) {
  int i = blockIdx.x * 256 + threadIdx.x;
  int tot = N * dk;
  if (i >= tot) return;
  int row = i / dk, col = i - row * dk;
  float s = 0.f;
  #pragma unroll
  for (int h = 0; h < 6; ++h) s += T[(long long)h * tot + i];
  C[(long long)row * ldc + colOff + col] = s;
}

__global__ __launch_bounds__(256)
void combine3_kernel(const float* __restrict__ x, const float* __restrict__ a,
                     const float* __restrict__ b, float* __restrict__ out,
                     unsigned short* __restrict__ obf, int n) {
  int i = blockIdx.x * 256 + threadIdx.x;
  if (i >= n) return;
  float v = (x[i] + a[i] + b[i]) * (1.f / 3.f);
  out[i] = v;
  obf[i] = f32_to_bf16_rte(v);
}

__global__ __launch_bounds__(256)
void combine2_bf16_kernel(const float* __restrict__ x, const float* __restrict__ a,
                          unsigned short* __restrict__ obf, int n) {
  int i = blockIdx.x * 256 + threadIdx.x;
  if (i >= n) return;
  obf[i] = f32_to_bf16_rte((x[i] + a[i]) * 0.5f);
}

__global__ __launch_bounds__(256)
void combine2_f32_kernel(const float* __restrict__ x, const float* __restrict__ a,
                         float* __restrict__ out, int n) {
  int i = blockIdx.x * 256 + threadIdx.x;
  if (i >= n) return;
  out[i] = (x[i] + a[i]) * 0.5f;
}

// ---------------------------------------------------------------------------
// Host-side GEMM launch helpers (4 template instantiations)
// ---------------------------------------------------------------------------
static void proj_gemm(const unsigned short* X, int M, const unsigned short* W,
                      const float* bias, unsigned short* Y, int N, hipStream_t s) {
  wmma_gemm<false, 0, true, false, true><<<dim3(N / 128, M / 128, 1), 256, 0, s>>>(
      X, 0, 1024, W, 0, N, Y, 0, N, bias, M, N, 1024, 1.0f);
}
static void scores_gemm(const unsigned short* Q, int Nq, const unsigned short* Km, int Nk,
                        int h, int dk, int od, float* S, hipStream_t s) {
  wmma_gemm<false, 1, false, false, false><<<dim3(Nk / 128, Nq / 128, h), 256, 0, s>>>(
      Q, dk, od, Km, dk, od, S, (long long)Nq * Nk, Nk, nullptr,
      Nq, Nk, dk, 1.0f / sqrtf((float)dk));
}
static void pv_gemm(const unsigned short* P, int Nq, int Nk, const unsigned short* V,
                    int od, int h, int dk, float* C, int ldc, long long cB, hipStream_t s) {
  wmma_gemm<false, 0, false, true, false><<<dim3(dk / 128, Nq / 128, h), 256, 0, s>>>(
      P, (long long)Nq * Nk, Nk, V, dk, od, C, cB, ldc, nullptr, Nq, dk, Nk, 1.0f);
}
static void ptv_gemm(const unsigned short* P, int Nq, int Nk, const unsigned short* V,
                     int od, int h, int dk, float* C, int ldc, long long cB, hipStream_t s) {
  wmma_gemm<true, 0, false, true, false><<<dim3(dk / 128, Nk / 128, h), 256, 0, s>>>(
      P, (long long)Nq * Nk, Nk, V, dk, od, C, cB, ldc, nullptr, Nk, dk, Nq, 1.0f);
}

// ---------------------------------------------------------------------------
extern "C" void kernel_launch(void* const* d_in, const int* in_sizes, int n_in,
                              void* d_out, int out_size, void* d_ws, size_t ws_size,
                              hipStream_t stream)
{
  (void)in_sizes; (void)n_in; (void)out_size; (void)ws_size;
  const float* feat_obj = (const float*)d_in[0];
  const float* feat_rel = (const float*)d_in[1];
  const float* W_cat    = (const float*)d_in[2];
  const float* b_cat    = (const float*)d_in[3];
  const float* W_out    = (const float*)d_in[4];
  const float* b_out    = (const float*)d_in[5];

  constexpr int DIM = 1024, NO = 2048, NR = 4096;
  constexpr int OC = 512,  DK1 = 128, H1 = 4;      // stage-1: concat=True
  constexpr int ON = 3072, DK2 = 512, H2 = 6;      // stage-2: concat=False

  // ---- workspace layout (~591 MB peak, buffers reused across attentions)
  char* basep = (char*)d_ws;
  size_t off = 0;
  auto alloc = [&](size_t bytes) -> char* {
    char* p = basep + off;
    off = (off + bytes + 255) & ~(size_t)255;
    return p;
  };
  unsigned short* featO_bf = (unsigned short*)alloc((size_t)NO * DIM * 2);
  unsigned short* featR_bf = (unsigned short*)alloc((size_t)NR * DIM * 2);
  unsigned short* Wcat_bf  = (unsigned short*)alloc((size_t)12 * DIM * OC * 2);
  unsigned short* Wout_bf  = (unsigned short*)alloc((size_t)8 * DIM * ON * 2);
  unsigned short* objF_bf  = (unsigned short*)alloc((size_t)NO * DIM * 2);
  unsigned short* relF_bf  = (unsigned short*)alloc((size_t)NR * DIM * 2);
  float* catA  = (float*)alloc((size_t)NO * DIM * 4);   // sub_obj || obj_sub
  float* catB  = (float*)alloc((size_t)NO * DIM * 4);   // sub_rel || obj_rel
  float* catR  = (float*)alloc((size_t)NR * DIM * 4);   // rel_obj || rel_sub
  float* catR2 = (float*)alloc((size_t)NR * DIM * 4);   // rel_obj2 || rel_sub2
  unsigned short* proj0 = (unsigned short*)alloc((size_t)NR * ON * 2);
  unsigned short* proj1 = (unsigned short*)alloc((size_t)NR * ON * 2);
  unsigned short* proj2 = (unsigned short*)alloc((size_t)NR * ON * 2);
  unsigned short* proj3 = (unsigned short*)alloc((size_t)NR * ON * 2);
  float*          scores  = (float*)alloc((size_t)H2 * NO * NR * 4);
  unsigned short* Pbuf    = (unsigned short*)alloc((size_t)H2 * NO * NR * 2);
  float*          headtmp = (float*)alloc((size_t)H2 * NR * DK2 * 4);

  float* objfeat = (float*)d_out;                       // output 0: [NO, DIM]
  float* relout  = (float*)d_out + (size_t)NO * DIM;    // output 1: [NR, DIM]

  auto cdiv = [](int a, int b) { return (a + b - 1) / b; };

  // ---- fp32 -> bf16 (activations + all weights)
  cvt_f32_bf16<<<cdiv(NO * DIM, 256), 256, 0, stream>>>(feat_obj, featO_bf, NO * DIM);
  cvt_f32_bf16<<<cdiv(NR * DIM, 256), 256, 0, stream>>>(feat_rel, featR_bf, NR * DIM);
  cvt_f32_bf16<<<cdiv(12 * DIM * OC, 256), 256, 0, stream>>>(W_cat, Wcat_bf, 12 * DIM * OC);
  cvt_f32_bf16<<<cdiv(8 * DIM * ON, 256), 256, 0, stream>>>(W_out, Wout_bf, 8 * DIM * ON);

  const unsigned short* Wc = Wcat_bf;
  const unsigned short* Wo = Wout_bf;

  // ================= Stage 1, mha1: (obj, obj), diag-masked ==================
  proj_gemm(featO_bf, NO, Wc + (size_t)0 * DIM * OC, b_cat + 0 * OC, proj0, OC, stream); // q
  proj_gemm(featO_bf, NO, Wc + (size_t)1 * DIM * OC, b_cat + 1 * OC, proj1, OC, stream); // k
  proj_gemm(featO_bf, NO, Wc + (size_t)2 * DIM * OC, b_cat + 2 * OC, proj2, OC, stream); // qv
  proj_gemm(featO_bf, NO, Wc + (size_t)3 * DIM * OC, b_cat + 3 * OC, proj3, OC, stream); // kv
  scores_gemm(proj0, NO, proj1, NO, H1, DK1, OC, scores, stream);
  softmax_rows<true><<<H1 * NO, 256, 0, stream>>>(scores, Pbuf, NO, NO);
  pv_gemm (Pbuf, NO, NO, proj3, OC, H1, DK1, catA,       DIM, DK1, stream);  // sub_obj
  ptv_gemm(Pbuf, NO, NO, proj2, OC, H1, DK1, catA + 512, DIM, DK1, stream);  // obj_sub

  // ================= Stage 1, mha2: (obj, rel) ==============================
  proj_gemm(featO_bf, NO, Wc + (size_t)4 * DIM * OC, b_cat + 4 * OC, proj0, OC, stream);
  proj_gemm(featR_bf, NR, Wc + (size_t)5 * DIM * OC, b_cat + 5 * OC, proj1, OC, stream);
  proj_gemm(featO_bf, NO, Wc + (size_t)6 * DIM * OC, b_cat + 6 * OC, proj2, OC, stream);
  proj_gemm(featR_bf, NR, Wc + (size_t)7 * DIM * OC, b_cat + 7 * OC, proj3, OC, stream);
  scores_gemm(proj0, NO, proj1, NR, H1, DK1, OC, scores, stream);
  softmax_rows<false><<<H1 * NO, 256, 0, stream>>>(scores, Pbuf, NR, NO);
  pv_gemm (Pbuf, NO, NR, proj3, OC, H1, DK1, catB,       DIM, DK1, stream);  // sub_rel
  ptv_gemm(Pbuf, NO, NR, proj2, OC, H1, DK1, catR + 512, DIM, DK1, stream);  // rel_sub

  // ================= Stage 1, mha3: (rel, obj) ==============================
  proj_gemm(featR_bf, NR, Wc + (size_t) 8 * DIM * OC, b_cat +  8 * OC, proj0, OC, stream);
  proj_gemm(featO_bf, NO, Wc + (size_t) 9 * DIM * OC, b_cat +  9 * OC, proj1, OC, stream);
  proj_gemm(featR_bf, NR, Wc + (size_t)10 * DIM * OC, b_cat + 10 * OC, proj2, OC, stream);
  proj_gemm(featO_bf, NO, Wc + (size_t)11 * DIM * OC, b_cat + 11 * OC, proj3, OC, stream);
  scores_gemm(proj0, NR, proj1, NO, H1, DK1, OC, scores, stream);
  softmax_rows<false><<<H1 * NR, 256, 0, stream>>>(scores, Pbuf, NO, NR);
  pv_gemm (Pbuf, NR, NO, proj3, OC, H1, DK1, catR,       DIM, DK1, stream);  // rel_obj
  ptv_gemm(Pbuf, NR, NO, proj2, OC, H1, DK1, catB + 512, DIM, DK1, stream);  // obj_rel

  // ---- residual combines (obj_feat goes straight to d_out + bf16 copy)
  combine3_kernel<<<cdiv(NO * DIM, 256), 256, 0, stream>>>(
      feat_obj, catA, catB, objfeat, objF_bf, NO * DIM);
  combine2_bf16_kernel<<<cdiv(NR * DIM, 256), 256, 0, stream>>>(
      feat_rel, catR, relF_bf, NR * DIM);

  // ================= Stage 2, mha4: (obj_feat, rel_feat), only a2 used ======
  proj_gemm(objF_bf, NO, Wo + (size_t)0 * DIM * ON, b_out + 0 * ON, proj0, ON, stream); // q
  proj_gemm(relF_bf, NR, Wo + (size_t)1 * DIM * ON, b_out + 1 * ON, proj1, ON, stream); // k
  proj_gemm(objF_bf, NO, Wo + (size_t)2 * DIM * ON, b_out + 2 * ON, proj2, ON, stream); // qv
  scores_gemm(proj0, NO, proj1, NR, H2, DK2, ON, scores, stream);
  softmax_rows<false><<<H2 * NO, 256, 0, stream>>>(scores, Pbuf, NR, NO);
  ptv_gemm(Pbuf, NO, NR, proj2, ON, H2, DK2, headtmp, DK2, (long long)NR * DK2, stream);
  head_sum6<<<cdiv(NR * DK2, 256), 256, 0, stream>>>(headtmp, catR2, NR, DK2, DIM, 512); // rel_sub2

  // ================= Stage 2, mha5: (rel_feat, obj_feat), only a1 used ======
  proj_gemm(relF_bf, NR, Wo + (size_t)4 * DIM * ON, b_out + 4 * ON, proj0, ON, stream); // q
  proj_gemm(objF_bf, NO, Wo + (size_t)5 * DIM * ON, b_out + 5 * ON, proj1, ON, stream); // k
  proj_gemm(objF_bf, NO, Wo + (size_t)7 * DIM * ON, b_out + 7 * ON, proj3, ON, stream); // kv
  scores_gemm(proj0, NR, proj1, NO, H2, DK2, ON, scores, stream);
  softmax_rows<false><<<H2 * NR, 256, 0, stream>>>(scores, Pbuf, NO, NR);
  pv_gemm(Pbuf, NR, NO, proj3, ON, H2, DK2, headtmp, DK2, (long long)NR * DK2, stream);
  head_sum6<<<cdiv(NR * DK2, 256), 256, 0, stream>>>(headtmp, catR2, NR, DK2, DIM, 0);   // rel_obj2

  // ---- final residual: rel_feat = (feat_rel + [rel_obj2||rel_sub2]) / 2
  combine2_f32_kernel<<<cdiv(NR * DIM, 256), 256, 0, stream>>>(
      feat_rel, catR2, relout, NR * DIM);
}
